// LHFA_76278619177511
// MI455X (gfx1250) — matrix operand order repaired and checked by reading, not compiled, not run on hardware
//
#include <hip/hip_runtime.h>

#define DEVI __device__ __forceinline__

typedef __bf16 bf16;
typedef __attribute__((ext_vector_type(16))) bf16 bf16x16;
typedef __attribute__((ext_vector_type(8)))  bf16 bf16x8;
typedef __attribute__((ext_vector_type(8)))  float f32x8;
typedef __attribute__((ext_vector_type(16))) float f32x16;

// ---------------- constants ----------------
#define BATCH 8
#define CIN   64
#define HH    128
#define WW    128
#define HWSZ  (HH * WW)      // 16384
#define DIM   256
#define HEADS 8
#define CHD   (DIM / HEADS)  // 32

// ---------------- bf16 helpers: native hardware converts ----------------
DEVI bf16 f2bf(float f)  { return (bf16)f; }     // v_cvt (RNE) on gfx1250
DEVI float bf2f(bf16 b)  { return (float)b; }
DEVI f32x8 zero8() { f32x8 z = {0.f,0.f,0.f,0.f,0.f,0.f,0.f,0.f}; return z; }

DEVI f32x8 wmma_bf16(bf16x16 a, bf16x16 b, f32x8 c) {
  return __builtin_amdgcn_wmma_f32_16x16x32_bf16(false, a, false, b, (short)0, c, false, false);
}

// A-frag: row-major A[M][K] (bf16), tile rows row0..row0+15, K = k0..k0+31.
// lane<16: m=lane, k = k0..k0+15; lane>=16: m=lane-16, k = k0+16..k0+31.
// Requires 16B alignment of (row stride, k0, chunk offsets) — true for all call sites.
DEVI bf16x16 load_a_frag(const bf16* __restrict__ A, int lda, int row0, int k0, int lane) {
  const bf16* r = A + (size_t)(row0 + (lane & 15)) * lda + k0 + ((lane >> 4) << 4);
  bf16x8 c0 = *(const bf16x8*)(r);
  bf16x8 c1 = *(const bf16x8*)(r + 8);
  return __builtin_shufflevector(c0, c1, 0,1,2,3,4,5,6,7, 8,9,10,11,12,13,14,15);
}

// B-frag from B^T storage: memory Bt[n][k] row-major (k contiguous), 16 contiguous k per lane.
DEVI bf16x16 load_bt_frag(const bf16* __restrict__ Bt, int ldk, int n0, int k0, int lane) {
  const bf16* r = Bt + (size_t)(n0 + (lane & 15)) * ldk + k0 + ((lane >> 4) << 4);
  bf16x8 c0 = *(const bf16x8*)(r);
  bf16x8 c1 = *(const bf16x8*)(r + 8);
  return __builtin_shufflevector(c0, c1, 0,1,2,3,4,5,6,7, 8,9,10,11,12,13,14,15);
}

// ---------------- K0: weight convert + accumulator zeroing ----------------
__global__ __launch_bounds__(256) void k_init(
    const float* __restrict__ Wq, const float* __restrict__ Wk, const float* __restrict__ Wv,
    const float* __restrict__ Wfus, const float* __restrict__ Wproj,
    bf16* __restrict__ wq, bf16* __restrict__ wk, bf16* __restrict__ wv,
    bf16* __restrict__ wfus, bf16* __restrict__ wproj,
    float* __restrict__ sumsq, float* __restrict__ attn_raw) {
  int i = blockIdx.x * blockDim.x + threadIdx.x;         // up to 65536
  if (i < DIM * CIN) { wq[i] = f2bf(Wq[i]); wk[i] = f2bf(Wk[i]); wv[i] = f2bf(Wv[i]); }
  if (i < DIM * 3 * CIN) wfus[i] = f2bf(Wfus[i]);
  if (i < DIM * DIM)     wproj[i] = f2bf(Wproj[i]);
  if (i < 2 * BATCH * DIM) sumsq[i] = 0.f;
  if (i < BATCH * HEADS * CHD * CHD) attn_raw[i] = 0.f;
}

// ---------------- K1: conv1x1 (WMMA) + dwconv3x3, emits bf16 + sumsq ----------------
#define HALO 18
#define HHP  (HALO * HALO)        // 324
#define NTIL ((HHP + 15) / 16)    // 21

__global__ __launch_bounds__(128) void k_qkv(
    const float* __restrict__ xin, const bf16* __restrict__ Wc,
    const float* __restrict__ Wdw, bf16* __restrict__ outq,
    float* __restrict__ sumsq) {
  const int lane = threadIdx.x & 31;
  const int wave = threadIdx.x >> 5;                 // 0..3
  const int blk  = blockIdx.x;                       // b*64 + sby*8 + sbx
  const int b    = blk >> 6;
  const int sby  = (blk >> 3) & 7;
  const int sbx  = blk & 7;

  // staging: [halo pixel][dim-in-tile]  -> WMMA result store is one b128 per lane
  __shared__ bf16 t_lds[4][HHP][16];

  const float* xb = xin + (size_t)b * CIN * HWSZ;
  const f32x16 vzero = {0.f,0.f,0.f,0.f,0.f,0.f,0.f,0.f,0.f,0.f,0.f,0.f,0.f,0.f,0.f,0.f};

  for (int dt4 = 0; dt4 < 4; ++dt4) {
    const int dt = wave * 4 + dt4;                   // dim tile 0..15
    const int o0 = dt * 16;
    bf16x16 a0 = load_a_frag(Wc, CIN, o0, 0, lane);
    bf16x16 a1 = load_a_frag(Wc, CIN, o0, 32, lane);

    // conv1x1 over the 18x18 halo, 21 pixel tiles of 16
    for (int nt = 0; nt < NTIL; ++nt) {
      const int n  = lane & 15;
      const int hp = nt * 16 + n;
      const int hr = hp / HALO, hc = hp % HALO;
      const int iy = sby * 16 - 1 + hr, ix = sbx * 16 - 1 + hc;
      const bool valid = (hp < HHP) && (iy >= 0) && (iy < HH) && (ix >= 0) && (ix < WW);
      // clamp to a safe address; zero out after (branch-free v_cndmask, no exec waltz)
      const long poff = valid ? ((long)iy * WW + ix) : 0;
      const float* xp = xb + poff;
      const int kb = (lane >> 4) << 4;
      f32x16 r0, r1;
#pragma unroll
      for (int i = 0; i < 16; ++i) {
        r0[i] = xp[(size_t)(kb + i) * HWSZ];
        r1[i] = xp[(size_t)(kb + i + 32) * HWSZ];
      }
      r0 = valid ? r0 : vzero;
      r1 = valid ? r1 : vzero;
      bf16x16 b0 = __builtin_convertvector(r0, bf16x16);
      bf16x16 b1 = __builtin_convertvector(r1, bf16x16);

      f32x8 acc = zero8();
      acc = wmma_bf16(a0, b0, acc);
      acc = wmma_bf16(a1, b1, acc);

      const int m0  = (lane >> 4) << 3;
      const int hpn = nt * 16 + (lane & 15);
      if (hpn < HHP) {
        bf16x8 cv = __builtin_convertvector(acc, bf16x8);
        *(bf16x8*)&t_lds[wave][hpn][m0] = cv;      // 16B aligned: hpn*32 + {0,16}
      }
    }

    // depthwise 3x3 (SAME, zero pad handled by zeroed halo), write bf16 + sumsq
    const bf16* tb = &t_lds[wave][0][0];
    for (int m = 0; m < 16; ++m) {
      const float* wd = Wdw + (size_t)(o0 + m) * 9;
      float w0 = wd[0], w1 = wd[1], w2 = wd[2];
      float w3 = wd[3], w4 = wd[4], w5 = wd[5];
      float w6 = wd[6], w7 = wd[7], w8 = wd[8];
      float s = 0.f;
      for (int pp = lane; pp < 256; pp += 32) {
        const int py = pp >> 4, px = pp & 15;
        const int h0 = ((py + 0) * HALO + px) * 16 + m;
        const int h1 = ((py + 1) * HALO + px) * 16 + m;
        const int h2 = ((py + 2) * HALO + px) * 16 + m;
        float acc = w0 * bf2f(tb[h0]) + w1 * bf2f(tb[h0 + 16]) + w2 * bf2f(tb[h0 + 32])
                  + w3 * bf2f(tb[h1]) + w4 * bf2f(tb[h1 + 16]) + w5 * bf2f(tb[h1 + 32])
                  + w6 * bf2f(tb[h2]) + w7 * bf2f(tb[h2 + 16]) + w8 * bf2f(tb[h2 + 32]);
        const int iy = sby * 16 + py, ix = sbx * 16 + px;
        outq[((size_t)b * DIM + o0 + m) * HWSZ + iy * WW + ix] = f2bf(acc);
        s += acc * acc;
      }
      if (sumsq) {
#pragma unroll
        for (int off = 16; off > 0; off >>= 1) s += __shfl_xor(s, off, 32);
        if (lane == 0) atomicAdd(&sumsq[b * DIM + o0 + m], s);
      }
    }
  }
}

// ---------------- K3: Gram matrices attn_raw = q k^T (K split, atomics) ----------------
#define KSPLIT 16
__global__ __launch_bounds__(32) void k_attn(
    const bf16* __restrict__ q, const bf16* __restrict__ k,
    float* __restrict__ attn_raw) {
  const int lane = threadIdx.x & 31;
  const int bh   = blockIdx.x;                // 0..63
  const int kc   = blockIdx.y;                // 0..KSPLIT-1
  const int b = bh >> 3, h = bh & 7;
  const bf16* qb = q + ((size_t)b * DIM + h * CHD) * HWSZ;
  const bf16* kb = k + ((size_t)b * DIM + h * CHD) * HWSZ;
  const int kbeg = kc * (HWSZ / KSPLIT);

  f32x8 acc00 = zero8(), acc01 = zero8(), acc10 = zero8(), acc11 = zero8();
#pragma unroll 2
  for (int p = kbeg; p < kbeg + HWSZ / KSPLIT; p += 32) {
    bf16x16 aq0 = load_a_frag(qb, HWSZ, 0,  p, lane);
    bf16x16 aq1 = load_a_frag(qb, HWSZ, 16, p, lane);
    bf16x16 bk0 = load_bt_frag(kb, HWSZ, 0,  p, lane);
    bf16x16 bk1 = load_bt_frag(kb, HWSZ, 16, p, lane);
    acc00 = wmma_bf16(aq0, bk0, acc00);
    acc01 = wmma_bf16(aq0, bk1, acc01);
    acc10 = wmma_bf16(aq1, bk0, acc10);
    acc11 = wmma_bf16(aq1, bk1, acc11);
  }
  float* ar = attn_raw + (size_t)bh * CHD * CHD;
  const int m0 = (lane >> 4) << 3;
  const int n  = lane & 15;
#pragma unroll
  for (int r = 0; r < 8; ++r) {
    atomicAdd(&ar[(0  + m0 + r) * CHD + 0  + n], acc00[r]);
    atomicAdd(&ar[(0  + m0 + r) * CHD + 16 + n], acc01[r]);
    atomicAdd(&ar[(16 + m0 + r) * CHD + 0  + n], acc10[r]);
    atomicAdd(&ar[(16 + m0 + r) * CHD + 16 + n], acc11[r]);
  }
}

// ---------------- K4: fold L2 norms + temperature, softmax, emit bf16 ----------------
__global__ __launch_bounds__(32) void k_softmax(
    const float* __restrict__ attn_raw, const float* __restrict__ ssq_q,
    const float* __restrict__ ssq_k, const float* __restrict__ temp,
    bf16* __restrict__ attn_bf) {
  const int lane = threadIdx.x & 31;
  const int bh = blockIdx.x;
  const int b = bh >> 3, h = bh & 7;
  __shared__ float rk_s[32];
  rk_s[lane] = 1.f / fmaxf(sqrtf(ssq_k[b * DIM + h * CHD + lane]), 1e-12f);
  __syncthreads();
  const float rq = 1.f / fmaxf(sqrtf(ssq_q[b * DIM + h * CHD + lane]), 1e-12f);
  const float t  = temp[h];
  const float* row = attn_raw + ((size_t)bh * CHD + lane) * CHD;
  float vals[CHD];
  float mx = -3.4e38f;
#pragma unroll
  for (int d = 0; d < CHD; ++d) {
    vals[d] = row[d] * rq * rk_s[d] * t;
    mx = fmaxf(mx, vals[d]);
  }
  float sum = 0.f;
#pragma unroll
  for (int d = 0; d < CHD; ++d) { vals[d] = __expf(vals[d] - mx); sum += vals[d]; }
  const float inv = 1.f / sum;
  bf16* orow = attn_bf + ((size_t)bh * CHD + lane) * CHD;
#pragma unroll
  for (int d = 0; d < CHD; ++d) orow[d] = f2bf(vals[d] * inv);
}

// ---------------- K5: out = Wproj @ (attn @ v) + Wfus @ concat(x,y,z) ----------------
__global__ __launch_bounds__(512) void k_out(
    const bf16* __restrict__ attn_bf, const bf16* __restrict__ v,
    const bf16* __restrict__ wproj, const bf16* __restrict__ wfus,
    const float* __restrict__ x, const float* __restrict__ y, const float* __restrict__ z,
    float* __restrict__ out) {
  const int lane = threadIdx.x & 31;
  const int wave = threadIdx.x >> 5;               // 0..15 (dim-tile / o-tile)
  const int blk  = blockIdx.x;                     // b*1024 + ptile
  const int b    = blk >> 10;
  const int p0   = (blk & 1023) * 16;

  __shared__ bf16 u_t[16][264];                    // u transposed: [pixel][dim]
  __shared__ bf16 conc[16][200];                   // concat(x,y,z) tile: [pixel][3C]

  // cooperative stage of concat inputs (bf16)
  for (int idx = threadIdx.x; idx < 16 * 3 * CIN; idx += 512) {
    const int p = idx / (3 * CIN), c = idx % (3 * CIN);
    const float* src = (c < CIN) ? x : (c < 2 * CIN) ? y : z;
    conc[p][c] = f2bf(src[((size_t)b * CIN + (c & (CIN - 1))) * HWSZ + p0 + p]);
  }

  // stage 1: u[dim-tile] = attn @ v  (one 16x16x32 WMMA per wave)
  {
    const int h  = wave >> 1;
    const int c0 = (wave & 1) * 16;
    bf16x16 a = load_a_frag(attn_bf + (size_t)(b * HEADS + h) * CHD * CHD, CHD, c0, 0, lane);
    const bf16* vb = v + ((size_t)b * DIM + h * CHD) * HWSZ + p0 + (lane & 15);
    const int kb = (lane >> 4) << 4;
    bf16x16 bb;
#pragma unroll
    for (int i = 0; i < 16; ++i) bb[i] = vb[(size_t)(kb + i) * HWSZ];
    f32x8 acc = zero8();
    acc = wmma_bf16(a, bb, acc);
    const int m0 = (lane >> 4) << 3;
    const int n  = lane & 15;
    bf16x8 cv = __builtin_convertvector(acc, bf16x8);
    *(bf16x8*)&u_t[n][wave * 16 + m0] = cv;        // 16B aligned: n*528 + wave*32 + {0,16}
  }
  __syncthreads();

  // stage 2: projection (K=256) + fusion (K=192) into one accumulator
  const int o0 = wave * 16;
  f32x8 acc = zero8();
#pragma unroll
  for (int k0 = 0; k0 < DIM; k0 += 32) {
    bf16x16 a  = load_a_frag(wproj, DIM, o0, k0, lane);
    bf16x16 bb = load_bt_frag(&u_t[0][0], 264, 0, k0, lane);
    acc = wmma_bf16(a, bb, acc);
  }
#pragma unroll
  for (int k0 = 0; k0 < 3 * CIN; k0 += 32) {
    bf16x16 a  = load_a_frag(wfus, 3 * CIN, o0, k0, lane);
    bf16x16 bb = load_bt_frag(&conc[0][0], 200, 0, k0, lane);
    acc = wmma_bf16(a, bb, acc);
  }
  const int m0 = (lane >> 4) << 3;
  const int n  = lane & 15;
  float* ob = out + ((size_t)b * DIM + o0) * HWSZ + p0 + n;
#pragma unroll
  for (int r = 0; r < 8; ++r) ob[(size_t)(m0 + r) * HWSZ] = acc[r];
}

// ---------------- launch ----------------
extern "C" void kernel_launch(void* const* d_in, const int* in_sizes, int n_in,
                              void* d_out, int out_size, void* d_ws, size_t ws_size,
                              hipStream_t stream) {
  (void)in_sizes; (void)n_in; (void)out_size; (void)ws_size;
  const float* x      = (const float*)d_in[0];
  const float* y      = (const float*)d_in[1];
  const float* z      = (const float*)d_in[2];
  const float* Wq     = (const float*)d_in[3];
  const float* Wq_dw  = (const float*)d_in[4];
  const float* Wk     = (const float*)d_in[5];
  const float* Wk_dw  = (const float*)d_in[6];
  const float* Wv     = (const float*)d_in[7];
  const float* Wv_dw  = (const float*)d_in[8];
  const float* Wproj  = (const float*)d_in[9];
  const float* Wfus   = (const float*)d_in[10];
  const float* temp   = (const float*)d_in[11];
  float* out = (float*)d_out;

  char* ws = (char*)d_ws;
  size_t off = 0;
  auto take = [&](size_t bytes) { char* p = ws + off; off = (off + bytes + 255) & ~(size_t)255; return p; };
  bf16*  wq_b   = (bf16*)take((size_t)DIM * CIN * 2);
  bf16*  wk_b   = (bf16*)take((size_t)DIM * CIN * 2);
  bf16*  wv_b   = (bf16*)take((size_t)DIM * CIN * 2);
  bf16*  wfus_b = (bf16*)take((size_t)DIM * 3 * CIN * 2);
  bf16*  wproj_b= (bf16*)take((size_t)DIM * DIM * 2);
  float* sumsq  = (float*)take((size_t)2 * BATCH * DIM * 4);   // [q | k]
  float* attn_r = (float*)take((size_t)BATCH * HEADS * CHD * CHD * 4);
  bf16*  attn_b = (bf16*)take((size_t)BATCH * HEADS * CHD * CHD * 2);
  bf16*  qbuf   = (bf16*)take((size_t)BATCH * DIM * HWSZ * 2);
  bf16*  kbuf   = (bf16*)take((size_t)BATCH * DIM * HWSZ * 2);
  bf16*  vbuf   = (bf16*)take((size_t)BATCH * DIM * HWSZ * 2);
  float* ssq_q  = sumsq;
  float* ssq_k  = sumsq + BATCH * DIM;

  k_init<<<256, 256, 0, stream>>>(Wq, Wk, Wv, Wfus, Wproj,
                                  wq_b, wk_b, wv_b, wfus_b, wproj_b, sumsq, attn_r);

  const int qkv_blocks = BATCH * (HH / 16) * (WW / 16);        // 512
  k_qkv<<<qkv_blocks, 128, 0, stream>>>(x, wq_b, Wq_dw, qbuf, ssq_q);
  k_qkv<<<qkv_blocks, 128, 0, stream>>>(y, wk_b, Wk_dw, kbuf, ssq_k);
  k_qkv<<<qkv_blocks, 128, 0, stream>>>(z, wv_b, Wv_dw, vbuf, nullptr);

  k_attn<<<dim3(BATCH * HEADS, KSPLIT), 32, 0, stream>>>(qbuf, kbuf, attn_r);
  k_softmax<<<BATCH * HEADS, 32, 0, stream>>>(attn_r, ssq_q, ssq_k, temp, attn_b);

  k_out<<<BATCH * (HWSZ / 16), 512, 0, stream>>>(attn_b, vbuf, wproj_b, wfus_b, x, y, z, out);
}
